// SDLoss_43215960932799
// MI455X (gfx1250) — matrix operand order repaired and checked by reading, not compiled, not run on hardware
//
#include <hip/hip_runtime.h>
#include <hip/hip_bf16.h>
#include <math.h>

typedef __attribute__((ext_vector_type(16))) __bf16 v16bf;
typedef __attribute__((ext_vector_type(8)))  float  v8f;

#define NEGV (-1.0e30f)

__device__ __forceinline__ float ladd(float x, float y) {
  float m = fmaxf(x, y);
  float d = fabsf(x - y);
  return m + log1pf(__expf(-d));
}

// ---------------------------------------------------------------------------
// Kernel 1: E = exp(trans) packed into WMMA B-operand tile layout (bf16).
// Tiles: [kb(8)][nt(16)][lane(32)][h(16)] ; element (k'=kb*32+k, c=nt*16+n):
//   lane = n + ((k&8)?16:0) ; halfword h = (k&7) | ((k&16)?8:0)   (ISA 7.12.2 B 32x16)
// ---------------------------------------------------------------------------
__global__ void prep_E_kernel(const float* __restrict__ trans,
                              __bf16* __restrict__ Eops) {
  int idx = blockIdx.x * blockDim.x + threadIdx.x; // 0..65535
  int kk = idx >> 8;   // source class c' (K dim)
  int c  = idx & 255;  // dest class    (N dim)
  float v = __expf(trans[idx]);
  int kb = kk >> 5, k = kk & 31;
  int nt = c >> 4,  n = c & 15;
  int lane = n + ((k & 8) ? 16 : 0);
  int h = (k & 7) | ((k & 16) ? 8 : 0);
  Eops[((kb * 16 + nt) * 32 + lane) * 16 + h] = (__bf16)v;
}

// A-operand (16x32 bf16, ISA 7.12.2) slot of element (m, c):
//   k=c&31, kb=c>>5, lane = m + ((k&8)?16:0), h = (k&7) | ((k&16)?8:0)
__device__ __forceinline__ int a_off(int m, int c) {
  int k = c & 31;
  int kb = c >> 5;
  int lw = m + ((k & 8) ? 16 : 0);
  int h  = (k & 7) | ((k & 16) ? 8 : 0);
  return (kb * 32 + lw) * 16 + h;   // halfword index within one buffer
}

// ---------------------------------------------------------------------------
// Kernel 2: denominator forward in linear space with periodic rescaling.
// Grid: B/16 blocks x 512 threads (16 waves). Wave w owns N-tile of 16 cols.
// Per step: alpha[16x256] x E[256,16] via 8x v_wmma_f32_16x16x32_bf16.
// logp loads are hoisted ahead of the WMMA chain (latency overlap);
// freeze (t >= in_len) is blended arithmetically so it cannot be
// if-converted into exec branches that would re-serialize the loads.
// ---------------------------------------------------------------------------
__global__ __launch_bounds__(512) void den_kernel(
    const float* __restrict__ logp, const float* __restrict__ startv,
    const int* __restrict__ in_raw, const __bf16* __restrict__ Eops,
    float* __restrict__ den_out, int T, int U) {
  const int C = 256;
  __shared__ __align__(32) __bf16 alphaA[2][8][32][16]; // double-buffered A-layout
  __shared__ float partial2[256];
  __shared__ float Lacc[16];
  __shared__ float invs[16];
  __shared__ int   ilen_s[16];

  const int tid   = threadIdx.x;
  const int lane  = tid & 31;
  const int wave  = tid >> 5;          // == nt (N tile)
  const int bbase = blockIdx.x * 16;
  const int Smin  = 2 * U + 1;

  if (tid < 16) {
    int il = in_raw[bbase + tid];
    il = il < Smin ? Smin : (il > T ? T : il);
    ilen_s[tid] = il;
    Lacc[tid] = 0.f;
  }
  // alpha(t=0) = exp(start + logp[:,0,:])   (L=0; first renorm at t=15)
  for (int j = 0; j < 8; ++j) {
    int e = tid * 8 + j;                  // 0..4095
    int m = e >> 8, c = e & 255;
    float v = __expf(startv[c] + logp[((size_t)(bbase + m) * T) * C + c]);
    ((__bf16*)alphaA)[a_off(m, c)] = (__bf16)v;
  }
  __syncthreads();

  const v16bf* Ev = (const v16bf*)Eops;
  const int nt = wave;
  const int nn = lane & 15;
  const int hi = lane >> 4;
  const int c_mine = nt * 16 + nn;

  // Per-lane loop-invariant state for the 8 owned D elements (m = r + 8*hi).
  int      ilen_r[8];
  unsigned aoff_r[8];                    // halfword offset within one buffer
  unsigned goff_r[8];                    // logp row offset (elements), m*T*C
  float    prev[8];
#pragma unroll
  for (int r = 0; r < 8; ++r) {
    int m = r + (hi ? 8 : 0);
    ilen_r[r] = ilen_s[m];
    aoff_r[r] = a_off(m, c_mine);
    goff_r[r] = (unsigned)m * (unsigned)T * (unsigned)C;
    prev[r] = (float)(((const __bf16*)alphaA)[aoff_r[r]]);
  }
  const float* lpbase = logp + (size_t)bbase * T * C + c_mine + C; // row t=1

  int buf = 0;
  for (int t = 1; t < T; ++t) {
    // Issue all 8 emission loads first: latency hides behind the WMMA chain.
    float lpv[8];
#pragma unroll
    for (int r = 0; r < 8; ++r) lpv[r] = lpbase[goff_r[r]];
    __builtin_prefetch(lpbase + goff_r[hi ? 7 : 0] + 8 * C, 0, 1);

    __bf16* acur = (__bf16*)alphaA + (size_t)buf * 4096;
    __bf16* anxt = (__bf16*)alphaA + (size_t)(buf ^ 1) * 4096;

    v8f acc = {0.f, 0.f, 0.f, 0.f, 0.f, 0.f, 0.f, 0.f};
#pragma unroll
    for (int kb = 0; kb < 8; ++kb) {
      v16bf a = *(const v16bf*)(acur + (kb * 32 + lane) * 16);
      v16bf b = Ev[(kb * 16 + nt) * 32 + lane];
      acc = __builtin_amdgcn_wmma_f32_16x16x32_bf16(
          false, a, false, b, (short)0, acc, false, false);
    }
#pragma unroll
    for (int r = 0; r < 8; ++r) {
      float prod = acc[r] * __expf(lpv[r]);
      float selF = (t >= ilen_r[r]) ? 1.0f : 0.0f; // cndmask, not a branch
      float nv = prod + selF * (prev[r] - prod);   // frozen ? prev : prod
      __bf16 w = (__bf16)nv;
      anxt[aoff_r[r]] = w;
      prev[r] = (float)w;
    }
    lpbase += C;
    __syncthreads();
    buf ^= 1;

    if ((t & 15) == 15 || t == T - 1) {  // periodic per-batch rescale
      if (tid < 256) {                   // row sums via b128 loads
        int kb = tid >> 5, lw = tid & 31;
        v16bf row = *(const v16bf*)((__bf16*)alphaA + (size_t)buf * 4096 +
                                    (kb * 32 + lw) * 16);
        float s = 0.f;
#pragma unroll
        for (int i = 0; i < 16; ++i) s += (float)row[i];
        partial2[tid] = s;               // batch = lw & 15
      }
      __syncthreads();
      if (tid < 16) {
        float tot = 0.f;
#pragma unroll
        for (int kb = 0; kb < 8; ++kb)
          tot += partial2[kb * 32 + tid] + partial2[kb * 32 + tid + 16];
        tot = fmaxf(tot, 1e-30f);
        Lacc[tid] += __logf(tot);
        invs[tid] = 1.f / tot;
      }
      __syncthreads();
      __bf16* ab = (__bf16*)alphaA + (size_t)buf * 4096;
#pragma unroll
      for (int r = 0; r < 8; ++r) {      // ownership-mapped rescale
        int m = r + (hi ? 8 : 0);
        __bf16 w = (__bf16)(prev[r] * invs[m]);
        ab[aoff_r[r]] = w;
        prev[r] = (float)w;
      }
      __syncthreads();
    }
  }
  // den[b] = L[b] + log(sum_c alpha[b,c])
  if (tid < 256) {
    int kb = tid >> 5, lw = tid & 31;
    v16bf row = *(const v16bf*)((__bf16*)alphaA + (size_t)buf * 4096 +
                                (kb * 32 + lw) * 16);
    float s = 0.f;
#pragma unroll
    for (int i = 0; i < 16; ++i) s += (float)row[i];
    partial2[tid] = s;
  }
  __syncthreads();
  if (tid < 16) {
    float tot = 0.f;
#pragma unroll
    for (int kb = 0; kb < 8; ++kb)
      tot += partial2[kb * 32 + tid] + partial2[kb * 32 + tid + 16];
    den_out[bbase + tid] = Lacc[tid] + __logf(fmaxf(tot, 1e-30f));
  }
}

// ---------------------------------------------------------------------------
// Kernel 3: CTC numerator forward (log domain), one workgroup per batch.
// ---------------------------------------------------------------------------
__global__ __launch_bounds__(512) void ctc_num_kernel(
    const float* __restrict__ logp, const int* __restrict__ targets,
    const int* __restrict__ in_raw, const int* __restrict__ tgt_raw,
    float* __restrict__ num_out, int T, int U) {
  const int C = 256;
  const int S = 2 * U + 1;
  if (S > 513) return;
  __shared__ int   ext_s[513];
  __shared__ int   skip_s[513];
  __shared__ float alpha_s[2][514];
  __shared__ float lprow[256];

  int b = blockIdx.x, tid = threadIdx.x;
  int ilen = in_raw[b];  ilen = ilen < S ? S : (ilen > T ? T : ilen);
  int tlen = tgt_raw[b]; tlen = tlen < 1 ? 1 : (tlen > U ? U : tlen);

  for (int s = tid; s < S; s += blockDim.x) {
    int lab = 0;
    if (s & 1) {
      lab = targets[b * U + (s >> 1)];
      lab = lab < 1 ? 1 : (lab > C - 1 ? C - 1 : lab);
    }
    ext_s[s] = lab;
  }
  __syncthreads();
  for (int s = tid; s < S; s += blockDim.x)
    skip_s[s] = (s >= 2) && (ext_s[s] != 0) && (ext_s[s] != ext_s[s - 2]);
  if (tid < C) lprow[tid] = logp[((size_t)b * T) * C + tid];
  __syncthreads();
  for (int s = tid; s < S; s += blockDim.x)
    alpha_s[0][s] = (s < 2) ? lprow[ext_s[s]] : NEGV;

  int cur = 0;
  for (int t = 1; t < T; ++t) {
    __syncthreads();
    if (tid < C) {
      __builtin_prefetch(&logp[((size_t)b * T + t + 4) * C + tid], 0, 1);
      lprow[tid] = logp[((size_t)b * T + t) * C + tid];
    }
    __syncthreads();
    int nxt = cur ^ 1;
    for (int s = tid; s < S; s += blockDim.x) {
      float a0 = alpha_s[cur][s];
      float a1 = (s >= 1) ? alpha_s[cur][s - 1] : NEGV;
      float a2 = skip_s[s] ? alpha_s[cur][s - 2] : NEGV;
      float nv = ladd(ladd(a0, a1), a2) + lprow[ext_s[s]];
      if (t >= ilen) nv = a0;                 // freeze past utterance end
      alpha_s[nxt][s] = nv;
    }
    cur = nxt;
  }
  __syncthreads();
  if (tid == 0) {
    int last = 2 * tlen;
    num_out[b] = ladd(alpha_s[cur][last], alpha_s[cur][last - 1]);
  }
}

// ---------------------------------------------------------------------------
// Kernel 4: tot = num - den; filter -inf; negated mean.
// ---------------------------------------------------------------------------
__global__ void finalize_kernel(const float* __restrict__ num,
                                const float* __restrict__ den,
                                float* __restrict__ out, int B) {
  if (threadIdx.x == 0 && blockIdx.x == 0) {
    float s = 0.f; int n = 0;
    for (int b = 0; b < B; ++b) {
      float tot = num[b] - den[b];
      if (tot > 0.5f * NEGV) { s += tot; n++; }
    }
    out[0] = -s / (float)(n < 1 ? 1 : n);
  }
}

extern "C" void kernel_launch(void* const* d_in, const int* in_sizes, int n_in,
                              void* d_out, int out_size, void* d_ws, size_t ws_size,
                              hipStream_t stream) {
  (void)n_in; (void)out_size; (void)ws_size;
  const float* logp   = (const float*)d_in[0];
  const int*   tgt    = (const int*)d_in[1];
  const int*   ilen   = (const int*)d_in[2];
  const int*   tlen   = (const int*)d_in[3];
  const float* trans  = (const float*)d_in[4];
  const float* startv = (const float*)d_in[5];

  const int B = in_sizes[2];                 // 32
  const int U = in_sizes[1] / B;             // 200
  const int C = in_sizes[5];                 // 256
  const int T = in_sizes[0] / (B * C);       // 2000

  __bf16* Eops = (__bf16*)d_ws;                                      // 128 KB
  float* den = (float*)((char*)d_ws + (size_t)C * C * sizeof(__bf16));
  float* num = den + B;

  prep_E_kernel<<<(C * C) / 256, 256, 0, stream>>>(trans, Eops);
  den_kernel<<<B / 16, 512, 0, stream>>>(logp, startv, ilen, Eops, den, T, U);
  ctc_num_kernel<<<B, 512, 0, stream>>>(logp, tgt, ilen, tlen, num, T, U);
  finalize_kernel<<<1, 32, 0, stream>>>(num, den, (float*)d_out, B);
}
